// AFMADEBlock_12232066859527
// MI455X (gfx1250) — compile-verified, matrix operand order checked by reading, not run on hardware
//
#include <hip/hip_runtime.h>
#include <cstdint>

// ---------------------------------------------------------------------------
// MADE inverse-autoregressive flow, D=16, H=1024, B=2048, 16 sequential iters.
// bf16 WMMA (v_wmma_f32_16x16x32_bf16) with f32 accumulation; weights
// weight-normalized+masked once per launch into bf16 workspace.
// Layer-1 GEMM uses the Tensor Data Mover (tensor_load_to_lds) to DMA padded
// A/B tiles into LDS with double buffering, synced via TENSORcnt.
// ---------------------------------------------------------------------------

#define DD 16
#define HH 1024
#define BB 2048

typedef __attribute__((ext_vector_type(16))) __bf16 v16bf;
typedef __attribute__((ext_vector_type(8)))  float  v8f;
typedef unsigned int  u32x4 __attribute__((ext_vector_type(4)));
typedef int           i32x8 __attribute__((ext_vector_type(8)));
typedef int           i32x4 __attribute__((ext_vector_type(4)));
typedef unsigned short u16;

#if defined(__HIP_DEVICE_COMPILE__) && __has_builtin(__builtin_amdgcn_tensor_load_to_lds) && __has_builtin(__builtin_amdgcn_s_wait_tensorcnt)
#define USE_TDM 1
#else
#define USE_TDM 0
#endif

union Frag {
    v16bf v;
    uint4 q[2];
};

__device__ __forceinline__ u16 f2bf(float f) {
    unsigned u = __float_as_uint(f);
    unsigned r = u + 0x7FFFu + ((u >> 16) & 1u);
    return (u16)(r >> 16);
}

__device__ __forceinline__ v8f vzero8() {
    v8f z;
#pragma unroll
    for (int i = 0; i < 8; ++i) z[i] = 0.0f;
    return z;
}

__device__ __forceinline__ v8f wmma_bf16(const Frag& a, const Frag& b, v8f c) {
    return __builtin_amdgcn_wmma_f32_16x16x32_bf16(
        /*neg_a=*/false, a.v, /*neg_b=*/false, b.v,
        /*c_mod=*/(short)0, c, /*reuse_a=*/false, /*reuse_b=*/false);
}

__device__ __forceinline__ float elu(float v) {
    return v > 0.0f ? v : (__expf(v) - 1.0f);
}

// ---------------------------------------------------------------------------
// Weight prep: w[o,i] = mask(o,i) * g[o] * v[o,i] / ||v[o,:]||   -> bf16
// layer 0: mask = deg_h[o] >= (i+1)         (O=H, I=D)
// layer 1: mask = deg_h[o] >= deg_h[i]      (O=H, I=H)
// layer 2: mask = (o+1)    >  deg_h[i]      (O=D, I=H)
// deg_h[k] = k % 15 + 1
// ---------------------------------------------------------------------------
__global__ void k_prep(const float* __restrict__ v, const float* __restrict__ g,
                       u16* __restrict__ w, int O, int I, int layer) {
    const int o = blockIdx.x;
    const int tid = threadIdx.x;
    __shared__ float red[256];
    float s = 0.0f;
    for (int i = tid; i < I; i += 256) {
        float t = v[(size_t)o * I + i];
        s += t * t;
    }
    red[tid] = s;
    __syncthreads();
    for (int st = 128; st > 0; st >>= 1) {
        if (tid < st) red[tid] += red[tid + st];
        __syncthreads();
    }
    const float scale = g[o] * rsqrtf(red[0]);
    const int dego = (layer == 2) ? (o + 1) : (o % 15 + 1);
    for (int i = tid; i < I; i += 256) {
        const int degi = (layer == 0) ? (i + 1) : (i % 15 + 1);
        const bool on = (layer == 2) ? (dego > degi) : (dego >= degi);
        const float t = on ? scale * v[(size_t)o * I + i] : 0.0f;
        w[(size_t)o * I + i] = f2bf(t);
    }
}

__global__ void k_zero(unsigned int* __restrict__ p, int n) {
    int i = blockIdx.x * blockDim.x + threadIdx.x;
    if (i < n) p[i] = 0u;
}

// ---------------------------------------------------------------------------
// Layer 0: h = ELU(y @ w0^T + b0).  K=16 (zero-padded to 32).
// grid (B/16, 1, 2 nets), block 256 (8 waves); wave -> 16x128 strip of N.
// ---------------------------------------------------------------------------
__global__ void __launch_bounds__(256)
k_l0(const u16* __restrict__ ybf,
     const u16* __restrict__ w0m, const u16* __restrict__ w0l,
     const float* __restrict__ b0m, const float* __restrict__ b0l,
     u16* __restrict__ hm, u16* __restrict__ hl) {
    const int net = blockIdx.z;
    const u16* __restrict__ w0 = net ? w0l : w0m;
    const float* __restrict__ b0 = net ? b0l : b0m;
    u16* __restrict__ hout = net ? hl : hm;

    const int tid = threadIdx.x;
    const int lane = tid & 31, wid = tid >> 5;
    const int lm = lane & 15, hk = lane >> 4;
    const int mBase = blockIdx.x * 16;

    Frag a;
    a.q[0] = *(const uint4*)(ybf + (size_t)(mBase + lm) * DD + 8 * hk);
    a.q[1] = make_uint4(0, 0, 0, 0);   // K = 16..31 padded with zeros

#pragma unroll
    for (int t = 0; t < 8; ++t) {
        const int nBase = wid * 128 + t * 16;
        Frag b;
        b.q[0] = *(const uint4*)(w0 + (size_t)(nBase + lm) * DD + 8 * hk);
        b.q[1] = make_uint4(0, 0, 0, 0);
        v8f c = vzero8();
        c = wmma_bf16(a, b, c);
#pragma unroll
        for (int r = 0; r < 8; ++r) {
            const int m = mBase + 8 * hk + r;
            const int n = nBase + lm;
            hout[(size_t)m * HH + n] = f2bf(elu(c[r] + b0[n]));
        }
    }
}

// ---------------------------------------------------------------------------
// Layer 1: h1 = ELU(h0 @ w1^T + b1).  M=2048, N=1024, K=1024.
// Block tile 128x128, 8 waves (2Mx4N), wave tile 64x32 = 4x2 WMMA accums.
// Staged K chunk = 64; LDS row stride 144B (64 data halfs + 8 pad halfs) ->
// conflict-free ds_load_b128 fragment fetches.
// TDM path: one wave DMAs each 128x64 tile (pad injected by TDM every 32
// DWORDs), double buffered, TENSORcnt-synced.
// ---------------------------------------------------------------------------
#define KC  64
#define LDR 72  // halfwords per LDS row = 144 bytes

#if USE_TDM
__device__ __forceinline__ void tdm_load_tile(const u16* gsrc, unsigned ldsByte,
                                              unsigned rowStrideElems,
                                              unsigned nRowsTensor) {
    unsigned long long ga = (unsigned long long)(size_t)gsrc;
    u32x4 g0;
    g0[0] = 1u;                                            // count=1, user mode
    g0[1] = ldsByte;                                       // lds_addr (bytes)
    g0[2] = (unsigned)(ga & 0xFFFFFFFFull);                // global_addr[31:0]
    g0[3] = (unsigned)((ga >> 32) & 0x1FFFFFFull)          // global_addr[56:32]
          | (2u << 30);                                    // type=2 ("image")
    i32x8 g1;
    g1[0] = (1 << 16)     // data_size = 2 bytes
          | (1 << 20)     // pad_enable
          | (4 << 22)     // pad_interval: 32 DWORDs (=128B tile row)
          | (3 << 25);    // pad_amount:   4 DWORDs (=16B)
    g1[1] = (int)((rowStrideElems & 0xFFFFu) << 16);       // tensor_dim0[15:0]
    g1[2] = (int)(((rowStrideElems >> 16) & 0xFFFFu)       // tensor_dim0[31:16]
          | ((nRowsTensor & 0xFFFFu) << 16));              // tensor_dim1[15:0]
    g1[3] = (int)(((nRowsTensor >> 16) & 0xFFFFu)          // tensor_dim1[31:16]
          | ((unsigned)KC << 16));                         // tile_dim0 = 64
    g1[4] = 128;                                           // tile_dim1=128, tile_dim2=0
    g1[5] = (int)rowStrideElems;                           // tensor_dim0_stride[31:0]
    g1[6] = 0;
    g1[7] = 0;
    i32x4 z4;
    z4[0] = z4[1] = z4[2] = z4[3] = 0;                     // 2D: groups 2/3 unused
    i32x8 z8;
#pragma unroll
    for (int i = 0; i < 8; ++i) z8[i] = 0;
    // 6-arg form (clang-23 / therock-10.0 headers): groups 0..3 + extra + cpol
    __builtin_amdgcn_tensor_load_to_lds(g0, g1, z4, z4, z8, 0);
}
#endif

__global__ void __launch_bounds__(256)
k_l1(const u16* __restrict__ hmi, const u16* __restrict__ hli,
     const u16* __restrict__ w1m, const u16* __restrict__ w1l,
     const float* __restrict__ b1m, const float* __restrict__ b1l,
     u16* __restrict__ hmo, u16* __restrict__ hlo) {
    const int net = blockIdx.z;
    const u16* __restrict__ Ain = net ? hli : hmi;
    const u16* __restrict__ W   = net ? w1l : w1m;
    const float* __restrict__ b1 = net ? b1l : b1m;
    u16* __restrict__ Hout = net ? hlo : hmo;

    __shared__ __align__(16) u16 sA[2][128 * LDR];
    __shared__ __align__(16) u16 sB[2][128 * LDR];

    const int tid = threadIdx.x;
    const int lane = tid & 31, wid = tid >> 5;
    const int lm = lane & 15, hk = lane >> 4;
    const int mBase = blockIdx.x * 128;
    const int nBase = blockIdx.y * 128;
    const int waveM = wid & 1;   // 0..1  (64 rows each)
    const int waveN = wid >> 1;  // 0..3  (32 cols each)
    const int nk = HH / KC;      // 16 staged chunks

    v8f acc[4][2];
#pragma unroll
    for (int mt = 0; mt < 4; ++mt)
#pragma unroll
        for (int nt = 0; nt < 2; ++nt) acc[mt][nt] = vzero8();

#if USE_TDM
    if (wid == 0) {  // prologue: fill buffer 0 with chunk 0 (A then B)
        tdm_load_tile(Ain + (size_t)mBase * HH, (unsigned)(size_t)&sA[0][0], HH, BB);
        tdm_load_tile(W   + (size_t)nBase * HH, (unsigned)(size_t)&sB[0][0], HH, HH);
    }
#endif

    for (int kc = 0; kc < nk; ++kc) {
        const int cur = kc & 1;
#if USE_TDM
        if (wid == 0) {
            if (kc + 1 < nk) {
                const size_t k1 = (size_t)(kc + 1) * KC;
                tdm_load_tile(Ain + (size_t)mBase * HH + k1,
                              (unsigned)(size_t)&sA[cur ^ 1][0], HH, BB);
                tdm_load_tile(W   + (size_t)nBase * HH + k1,
                              (unsigned)(size_t)&sB[cur ^ 1][0], HH, HH);
                __builtin_amdgcn_s_wait_tensorcnt(2);  // chunk kc (A+B) done
            } else {
                __builtin_amdgcn_s_wait_tensorcnt(0);
            }
        }
        __syncthreads();
#else
        // fallback: cooperative staging through VGPRs
        {
            const int k0 = kc * KC;
#pragma unroll
            for (int c = tid; c < 1024; c += 256) {
                const int r = c >> 3, cc = c & 7;
                *(uint4*)&sA[cur][r * LDR + cc * 8] =
                    *(const uint4*)(Ain + (size_t)(mBase + r) * HH + k0 + cc * 8);
                *(uint4*)&sB[cur][r * LDR + cc * 8] =
                    *(const uint4*)(W + (size_t)(nBase + r) * HH + k0 + cc * 8);
            }
        }
        __syncthreads();
#endif
        const u16* bufA = &sA[cur][0];
        const u16* bufB = &sB[cur][0];
#pragma unroll
        for (int kk = 0; kk < 2; ++kk) {
            const int kb = kk * 32;
            Frag af[4], bf[2];
#pragma unroll
            for (int mt = 0; mt < 4; ++mt) {
                const int r = waveM * 64 + mt * 16 + lm;
                af[mt].q[0] = *(const uint4*)&bufA[r * LDR + kb + 8 * hk];
                af[mt].q[1] = *(const uint4*)&bufA[r * LDR + kb + 16 + 8 * hk];
            }
#pragma unroll
            for (int nt = 0; nt < 2; ++nt) {
                const int r = waveN * 32 + nt * 16 + lm;
                bf[nt].q[0] = *(const uint4*)&bufB[r * LDR + kb + 8 * hk];
                bf[nt].q[1] = *(const uint4*)&bufB[r * LDR + kb + 16 + 8 * hk];
            }
#pragma unroll
            for (int mt = 0; mt < 4; ++mt)
#pragma unroll
                for (int nt = 0; nt < 2; ++nt)
                    acc[mt][nt] = wmma_bf16(af[mt], bf[nt], acc[mt][nt]);
        }
        __syncthreads();
    }

    // ---- epilogue: bias + ELU, store bf16 ----
#pragma unroll
    for (int mt = 0; mt < 4; ++mt)
#pragma unroll
        for (int nt = 0; nt < 2; ++nt)
#pragma unroll
            for (int r = 0; r < 8; ++r) {
                const int m = mBase + waveM * 64 + mt * 16 + 8 * hk + r;
                const int n = nBase + waveN * 32 + nt * 16 + lm;
                Hout[(size_t)m * HH + n] = f2bf(elu(acc[mt][nt][r] + b1[n]));
            }
}

// ---------------------------------------------------------------------------
// Layer 2 + flow update: mu = h_mu@w2m^T + b2m ; ls = 0.5*(h_lv@w2l^T + b2l)
// y <- (x - mu)/(exp(ls)+eps).  On last iter also write d_out (y, sum_n ls).
// grid (B/64), block 128 (4 waves); wave -> one 16x16 output tile, K=1024.
// ---------------------------------------------------------------------------
__global__ void __launch_bounds__(128)
k_l2(const u16* __restrict__ hm, const u16* __restrict__ hl,
     const u16* __restrict__ w2m, const u16* __restrict__ w2l,
     const float* __restrict__ b2m, const float* __restrict__ b2l,
     const float* __restrict__ x,
     u16* __restrict__ ybf, float* __restrict__ yout,
     float* __restrict__ lsout, int last) {
    const int tid = threadIdx.x;
    const int lane = tid & 31, wid = tid >> 5;
    const int lm = lane & 15, hk = lane >> 4;
    const int mB = blockIdx.x * 64 + wid * 16;

    v8f cm = vzero8(), cl = vzero8();
    const u16* am_row = hm + (size_t)(mB + lm) * HH;
    const u16* al_row = hl + (size_t)(mB + lm) * HH;
    const u16* bm_row = w2m + (size_t)lm * HH;
    const u16* bl_row = w2l + (size_t)lm * HH;

    for (int k0 = 0; k0 < HH; k0 += 32) {
        Frag am, bm, al, bl;
        am.q[0] = *(const uint4*)(am_row + k0 + 8 * hk);
        am.q[1] = *(const uint4*)(am_row + k0 + 16 + 8 * hk);
        bm.q[0] = *(const uint4*)(bm_row + k0 + 8 * hk);
        bm.q[1] = *(const uint4*)(bm_row + k0 + 16 + 8 * hk);
        cm = wmma_bf16(am, bm, cm);
        al.q[0] = *(const uint4*)(al_row + k0 + 8 * hk);
        al.q[1] = *(const uint4*)(al_row + k0 + 16 + 8 * hk);
        bl.q[0] = *(const uint4*)(bl_row + k0 + 8 * hk);
        bl.q[1] = *(const uint4*)(bl_row + k0 + 16 + 8 * hk);
        cl = wmma_bf16(al, bl, cl);
    }

#pragma unroll
    for (int r = 0; r < 8; ++r) {
        const int m = mB + 8 * hk + r;
        const float mu = cm[r] + b2m[lm];
        const float ls = 0.5f * (cl[r] + b2l[lm]);
        const float xv = x[(size_t)m * DD + lm];
        const float yv = (xv - mu) / (__expf(ls) + 1e-12f);
        ybf[(size_t)m * DD + lm] = f2bf(yv);
        if (last) {
            yout[(size_t)m * DD + lm] = yv;
            float s = ls;  // sum over n (16 lanes of same half)
            s += __shfl_xor(s, 1, 32);
            s += __shfl_xor(s, 2, 32);
            s += __shfl_xor(s, 4, 32);
            s += __shfl_xor(s, 8, 32);
            if (lm == 0) lsout[m] = s;
        }
    }
}

// ---------------------------------------------------------------------------
extern "C" void kernel_launch(void* const* d_in, const int* in_sizes, int n_in,
                              void* d_out, int out_size, void* d_ws, size_t ws_size,
                              hipStream_t stream) {
    (void)in_sizes; (void)n_in; (void)out_size; (void)ws_size;
    const float* x     = (const float*)d_in[0];
    const float* mu_v0 = (const float*)d_in[1];
    const float* mu_g0 = (const float*)d_in[2];
    const float* mu_b0 = (const float*)d_in[3];
    const float* mu_v1 = (const float*)d_in[4];
    const float* mu_g1 = (const float*)d_in[5];
    const float* mu_b1 = (const float*)d_in[6];
    const float* mu_v2 = (const float*)d_in[7];
    const float* mu_g2 = (const float*)d_in[8];
    const float* mu_b2 = (const float*)d_in[9];
    const float* lv_v0 = (const float*)d_in[10];
    const float* lv_g0 = (const float*)d_in[11];
    const float* lv_b0 = (const float*)d_in[12];
    const float* lv_v1 = (const float*)d_in[13];
    const float* lv_g1 = (const float*)d_in[14];
    const float* lv_b1 = (const float*)d_in[15];
    const float* lv_v2 = (const float*)d_in[16];
    const float* lv_g2 = (const float*)d_in[17];
    const float* lv_b2 = (const float*)d_in[18];

    char* base = (char*)d_ws;
    size_t off = 0;
    auto wsAlloc = [&](size_t bytes) -> void* {
        void* p = (void*)(base + off);
        off += (bytes + 255) & ~(size_t)255;
        return p;
    };
    u16* w0m = (u16*)wsAlloc((size_t)HH * DD * 2);
    u16* w1m = (u16*)wsAlloc((size_t)HH * HH * 2);
    u16* w2m = (u16*)wsAlloc((size_t)DD * HH * 2);
    u16* w0l = (u16*)wsAlloc((size_t)HH * DD * 2);
    u16* w1l = (u16*)wsAlloc((size_t)HH * HH * 2);
    u16* w2l = (u16*)wsAlloc((size_t)DD * HH * 2);
    u16* hm0 = (u16*)wsAlloc((size_t)BB * HH * 2);
    u16* hm1 = (u16*)wsAlloc((size_t)BB * HH * 2);
    u16* hl0 = (u16*)wsAlloc((size_t)BB * HH * 2);
    u16* hl1 = (u16*)wsAlloc((size_t)BB * HH * 2);
    u16* ybf = (u16*)wsAlloc((size_t)BB * DD * 2);

    float* yout  = (float*)d_out;
    float* lsout = yout + (size_t)BB * DD;

    // weight normalization + masking -> bf16 (constant across the 16 iters)
    k_prep<<<dim3(HH), 256, 0, stream>>>(mu_v0, mu_g0, w0m, HH, DD, 0);
    k_prep<<<dim3(HH), 256, 0, stream>>>(mu_v1, mu_g1, w1m, HH, HH, 1);
    k_prep<<<dim3(DD), 256, 0, stream>>>(mu_v2, mu_g2, w2m, DD, HH, 2);
    k_prep<<<dim3(HH), 256, 0, stream>>>(lv_v0, lv_g0, w0l, HH, DD, 0);
    k_prep<<<dim3(HH), 256, 0, stream>>>(lv_v1, lv_g1, w1l, HH, HH, 1);
    k_prep<<<dim3(DD), 256, 0, stream>>>(lv_v2, lv_g2, w2l, DD, HH, 2);

    // y = 0
    {
        int n = BB * DD / 2;  // as u32
        k_zero<<<dim3((n + 255) / 256), 256, 0, stream>>>((unsigned int*)ybf, n);
    }

    for (int it = 0; it < DD; ++it) {
        k_l0<<<dim3(BB / 16, 1, 2), 256, 0, stream>>>(ybf, w0m, w0l, mu_b0, lv_b0, hm0, hl0);
        k_l1<<<dim3(BB / 128, HH / 128, 2), 256, 0, stream>>>(hm0, hl0, w1m, w1l,
                                                              mu_b1, lv_b1, hm1, hl1);
        k_l2<<<dim3(BB / 64), 128, 0, stream>>>(hm1, hl1, w2m, w2l, mu_b2, lv_b2,
                                                x, ybf, yout, lsout,
                                                (it == DD - 1) ? 1 : 0);
    }
}
